// _LSTMNet_30623116821294
// MI455X (gfx1250) — matrix operand order repaired and checked by reading, not compiled, not run on hardware
//
#include <hip/hip_runtime.h>

// ---------------------------------------------------------------------------
// LSTM (B=2048, T=512, H=64) + MLP head, tuned for gfx1250 (MI455X, wave32).
//
// Strategy:
//  * 128 workgroups, one per 16-row batch tile; 4 waves per WG, one per gate.
//  * Recurrent matmul h @ W_hh^T done with v_wmma_f32_16x16x32_f16:
//    per step, per wave: 4 column tiles x 2 k-chunks = 8 WMMAs.
//  * W_hh pre-swizzled once into WMMA B-fragment layout in LDS (f16, 32KB).
//  * h kept in LDS (f16, padded stride), c kept in registers (f32).
//  * Gate pre-activations exchanged via padded f32 LDS + 2 barriers/step.
//  * x streamed in 16-step LDS chunks; bias + x*W_ih folded into WMMA C init.
// ---------------------------------------------------------------------------

typedef __attribute__((ext_vector_type(16))) _Float16 v16h;
typedef __attribute__((ext_vector_type(8)))  _Float16 v8h;
typedef __attribute__((ext_vector_type(8)))  float    v8f;

#define B_   2048
#define T_   512
#define H_   64
#define H2_  16
#define NTHREADS 128
#define XC   16            // x timestep chunk cached in LDS
#define HSTR 72            // padded h row stride (halves): 144B, 16B aligned

union V16U { v16h v; v8h h[2]; };

__device__ __forceinline__ float fast_sigmoid(float x) {
    return 1.0f / (1.0f + __expf(-x));
}
__device__ __forceinline__ float fast_tanh(float x) {
    // tanh(x) = 1 - 2/(exp(2x)+1)
    return 1.0f - 2.0f / (__expf(2.0f * x) + 1.0f);
}

__global__ __launch_bounds__(NTHREADS)
void lstm_wmma_kernel(const float* __restrict__ x,
                      const float* __restrict__ W_ih,
                      const float* __restrict__ W_hh,
                      const float* __restrict__ b_ih,
                      const float* __restrict__ b_hh,
                      const float* __restrict__ fc1_w,
                      const float* __restrict__ fc1_b,
                      const float* __restrict__ fc2_w,
                      const float* __restrict__ fc2_b,
                      float* __restrict__ out)
{
    // W_hh in WMMA B-fragment layout: [tile(16)][chunk(2)][lane(32)][e(16)] f16
    __shared__ _Float16 wfrag[16 * 2 * 32 * 16];          // 32 KB
    __shared__ float    gacc[4][16][65];                  // gate pre-acts, padded
    __shared__ _Float16 hlds[16][HSTR];                   // h state, f16
    __shared__ float    xlds[16][XC];                     // x chunk

    const int tid   = threadIdx.x;
    const int wave  = tid >> 5;       // gate id: 0=i 1=f 2=g 3=o
    const int lane  = tid & 31;
    const int lo    = lane & 15;
    const int hi    = lane >> 4;
    const int rows0 = blockIdx.x * 16;

    // ---- one-time: swizzle W_hh (f32 -> f16) into B-fragment layout ----
    // B[k, n] for output col (16*tile + (l&15)) is W_hh[col, k];
    // lane l holds K = chunk*32 + (l>>4)*16 + e  at its column.
    for (int idx = tid; idx < 16 * 2 * 32 * 16; idx += NTHREADS) {
        int e     = idx & 15;
        int l     = (idx >> 4) & 31;
        int chunk = (idx >> 9) & 1;
        int tile  = idx >> 10;
        int col   = tile * 16 + (l & 15);
        int k     = chunk * 32 + (l >> 4) * 16 + e;
        wfrag[idx] = (_Float16)W_hh[col * H_ + k];
    }
    // zero h state
    for (int idx = tid; idx < 16 * HSTR; idx += NTHREADS)
        (&hlds[0][0])[idx] = (_Float16)0.0f;

    // per-wave constants: fused bias and W_ih for this gate's 4 column tiles
    float biasv[4], wihv[4];
    #pragma unroll
    for (int ti = 0; ti < 4; ++ti) {
        int col   = wave * 64 + ti * 16 + lo;
        biasv[ti] = b_ih[col] + b_hh[col];
        wihv[ti]  = W_ih[col];
    }

    float c[8];
    #pragma unroll
    for (int r = 0; r < 8; ++r) c[r] = 0.0f;

    __syncthreads();

    for (int t = 0; t < T_; ++t) {
        if ((t & (XC - 1)) == 0) {
            // refresh x chunk: 16 rows x XC steps, coalesced along t
            for (int idx = tid; idx < 16 * XC; idx += NTHREADS) {
                int m  = idx >> 4;
                int tt = idx & (XC - 1);
                xlds[m][tt] = x[(rows0 + m) * T_ + t + tt];
            }
            __syncthreads();
        }
        const int tt = t & (XC - 1);

        // ---- A fragments (16x32 f16) gathered from h in LDS ----
        // rows M=lane&15, K = kb..kb+15 (+32 for a1), kb = (lane>=16)?16:0
        V16U a0, a1;
        {
            const _Float16* hp = &hlds[lo][0];
            const int kb = hi * 16;
            a0.h[0] = *(const v8h*)(hp + kb);
            a0.h[1] = *(const v8h*)(hp + kb + 8);
            a1.h[0] = *(const v8h*)(hp + kb + 32);
            a1.h[1] = *(const v8h*)(hp + kb + 40);
        }
        // x values for this lane's 8 output rows (broadcast reads)
        float xv[8];
        #pragma unroll
        for (int r = 0; r < 8; ++r) xv[r] = xlds[r + hi * 8][tt];

        // ---- 4 column tiles of this gate: C-init + 2 WMMAs each ----
        #pragma unroll
        for (int ti = 0; ti < 4; ++ti) {
            const int tile = wave * 4 + ti;
            const _Float16* wp = &wfrag[(tile * 2) * 512 + lane * 16];
            V16U b0, b1;
            b0.h[0] = *(const v8h*)(wp);
            b0.h[1] = *(const v8h*)(wp + 8);
            b1.h[0] = *(const v8h*)(wp + 512);
            b1.h[1] = *(const v8h*)(wp + 520);

            v8f acc;
            #pragma unroll
            for (int r = 0; r < 8; ++r) acc[r] = fmaf(xv[r], wihv[ti], biasv[ti]);

            acc = __builtin_amdgcn_wmma_f32_16x16x32_f16(
                      false, a0.v, false, b0.v, (short)0, acc, false, false);
            acc = __builtin_amdgcn_wmma_f32_16x16x32_f16(
                      false, a1.v, false, b1.v, (short)0, acc, false, false);

            // publish gate pre-activations (C-layout: row = r + hi*8, col = lane&15)
            #pragma unroll
            for (int r = 0; r < 8; ++r)
                gacc[wave][r + hi * 8][ti * 16 + lo] = acc[r];
        }
        __syncthreads();

        // ---- elementwise c/h update: wave w owns hidden cols [16w, 16w+16) ----
        {
            const int col = wave * 16 + lo;
            #pragma unroll
            for (int r = 0; r < 8; ++r) {
                int m    = r + hi * 8;
                float ig = fast_sigmoid(gacc[0][m][col]);
                float fg = fast_sigmoid(gacc[1][m][col]);
                float gg = fast_tanh   (gacc[2][m][col]);
                float og = fast_sigmoid(gacc[3][m][col]);
                c[r]     = fg * c[r] + ig * gg;
                float hv = og * fast_tanh(c[r]);
                hlds[m][col] = (_Float16)hv;
            }
        }
        __syncthreads();
    }

    // ---- MLP head on final h: fc1 (WMMA, 16x16 out, K=64) -> relu -> fc2 ----
    if (wave == 0) {
        V16U a0, a1;
        {
            const _Float16* hp = &hlds[lo][0];
            const int kb = hi * 16;
            a0.h[0] = *(const v8h*)(hp + kb);
            a0.h[1] = *(const v8h*)(hp + kb + 8);
            a1.h[0] = *(const v8h*)(hp + kb + 32);
            a1.h[1] = *(const v8h*)(hp + kb + 40);
        }
        // fc1_w^T B-fragments straight from global (one-time, tiny)
        V16U b0, b1;
        {
            const int n  = lo;
            const int kb = hi * 16;
            #pragma unroll
            for (int e = 0; e < 16; ++e) {
                b0.v[e] = (_Float16)fc1_w[n * H_ + kb + e];
                b1.v[e] = (_Float16)fc1_w[n * H_ + 32 + kb + e];
            }
        }
        v8f acc;
        {
            float bias = fc1_b[lo];
            #pragma unroll
            for (int r = 0; r < 8; ++r) acc[r] = bias;
        }
        acc = __builtin_amdgcn_wmma_f32_16x16x32_f16(
                  false, a0.v, false, b0.v, (short)0, acc, false, false);
        acc = __builtin_amdgcn_wmma_f32_16x16x32_f16(
                  false, a1.v, false, b1.v, (short)0, acc, false, false);

        // fc2: per-row dot over 16 cols spread across a 16-lane group
        const float w2 = fc2_w[lo];
        const float b2 = fc2_b[0];
        #pragma unroll
        for (int r = 0; r < 8; ++r) {
            float z = acc[r] > 0.0f ? acc[r] : 0.0f;
            float v = z * w2;
            v += __shfl_xor(v, 1, 32);
            v += __shfl_xor(v, 2, 32);
            v += __shfl_xor(v, 4, 32);
            v += __shfl_xor(v, 8, 32);
            if (lo == 0) out[rows0 + hi * 8 + r] = v + b2;
        }
    }
}

extern "C" void kernel_launch(void* const* d_in, const int* in_sizes, int n_in,
                              void* d_out, int out_size, void* d_ws, size_t ws_size,
                              hipStream_t stream) {
    const float* x     = (const float*)d_in[0];
    const float* W_ih  = (const float*)d_in[1];
    const float* W_hh  = (const float*)d_in[2];
    const float* b_ih  = (const float*)d_in[3];
    const float* b_hh  = (const float*)d_in[4];
    const float* fc1_w = (const float*)d_in[5];
    const float* fc1_b = (const float*)d_in[6];
    const float* fc2_w = (const float*)d_in[7];
    const float* fc2_b = (const float*)d_in[8];
    float* out = (float*)d_out;

    dim3 grid(B_ / 16);        // 128 workgroups (one per 16-row batch tile)
    dim3 block(NTHREADS);      // 4 waves: one per LSTM gate
    lstm_wmma_kernel<<<grid, block, 0, stream>>>(
        x, W_ih, W_hh, b_ih, b_hh, fc1_w, fc1_b, fc2_w, fc2_b, out);
}
